// GraphTransformer_38388417692072
// MI455X (gfx1250) — compile-verified, matrix operand order checked by reading, not run on hardware
//
#include <hip/hip_runtime.h>
#include <math.h>

typedef __attribute__((ext_vector_type(16))) __bf16 v16bf;
typedef __attribute__((ext_vector_type(8)))  __bf16 v8bf;
typedef __attribute__((ext_vector_type(4)))  __bf16 v4bf;
typedef __attribute__((ext_vector_type(8)))  float  v8f;

#define H_HEADS 4

// ---------------------------------------------------------------------------
// WMMA GEMM: C[M x Nc] = A_bf16[M x KC] * Bt_bf16[Nc x KC]^T + bias
// Block = 256 threads = 8 waves; each wave computes a 16 x (NT*16) strip of C
// so the A stream is read exactly once (B fragments are tiny, cache-resident).
// KC (64 or 128) is a template parameter -> K loop fully unrolls, and all NT
// B fragments of a k-step are staged in distinct registers before the wmma
// burst so loads clause up and waits stagger instead of serializing.
// A-fragment layout per CDNA5 ISA (16-bit A 16x32): lanes 0-15 row M=lid hold
// K {0..7,16..23}; lanes 16-31 hold K {8..15,24..31}. B mirrored (col-major).
// ---------------------------------------------------------------------------
template <int NT, int KC>
__global__ __launch_bounds__(256) void k_gemm_bf16_wmma(
    const __bf16* __restrict__ A, const __bf16* __restrict__ Bt,
    const float* __restrict__ bias,
    float* __restrict__ Cf, __bf16* __restrict__ Cb,
    int M)
{
    const int Nc    = NT * 16;
    const int lane  = threadIdx.x & 31;
    const int wave  = threadIdx.x >> 5;
    const int tileM = blockIdx.x * 8 + wave;
    if (tileM * 16 >= M) return;
    const int half = lane >> 4;      // 0 or 1
    const int lid  = lane & 15;
    const int off  = half * 8;

    const __bf16* __restrict__ arow = A  + (size_t)(tileM * 16 + lid) * KC + off;
    const __bf16* __restrict__ bcol = Bt + (size_t)lid * KC + off;

    v8f acc[NT];
#pragma unroll
    for (int t = 0; t < NT; ++t) acc[t] = {};

#pragma unroll
    for (int k0 = 0; k0 < KC; k0 += 32) {
        // A fragment for this k-step
        v8bf a0 = *(const v8bf*)(arow + k0);
        v8bf a1 = *(const v8bf*)(arow + k0 + 16);
        v16bf av;
#pragma unroll
        for (int i = 0; i < 8; ++i) { av[i] = a0[i]; av[8 + i] = a1[i]; }

        // Stage ALL B fragments of this k-step first (distinct registers) so
        // the loads clause and the wmma burst overlaps the tail of the loads.
        v16bf bfrag[NT];
#pragma unroll
        for (int t = 0; t < NT; ++t) {
            const __bf16* bp = bcol + (size_t)t * 16 * KC + k0;
            v8bf b0 = *(const v8bf*)bp;
            v8bf b1 = *(const v8bf*)(bp + 16);
#pragma unroll
            for (int i = 0; i < 8; ++i) { bfrag[t][i] = b0[i]; bfrag[t][8 + i] = b1[i]; }
        }

#pragma unroll
        for (int t = 0; t < NT; ++t)
            acc[t] = __builtin_amdgcn_wmma_f32_16x16x32_bf16(
                         false, av, false, bfrag[t], (short)0, acc[t], false, false);
    }

#pragma unroll
    for (int t = 0; t < NT; ++t) {
        const int col = t * 16 + lid;
        const float bval = bias ? bias[col] : 0.0f;
#pragma unroll
        for (int r = 0; r < 8; ++r) {
            const int row = tileM * 16 + r + half * 8;  // C layout: VGPR r -> M=r / M=r+8
            const float v = acc[t][r] + bval;
            if (Cf) Cf[(size_t)row * Nc + col] = v;
            else    Cb[(size_t)row * Nc + col] = (__bf16)v;
        }
    }
}

// ---------------------------------------------------------------------------
// Small prep kernels
// ---------------------------------------------------------------------------
__global__ void k_f32_to_bf16(const float* __restrict__ in, __bf16* __restrict__ out, size_t n) {
    size_t i = (size_t)blockIdx.x * blockDim.x + threadIdx.x;
    if (i < n) out[i] = (__bf16)in[i];
}

__global__ void k_transpose_bf16(const float* __restrict__ W, __bf16* __restrict__ Wt, int fi, int fo) {
    int idx = blockIdx.x * blockDim.x + threadIdx.x;
    if (idx >= fi * fo) return;
    int i = idx / fo, j = idx % fo;
    Wt[(size_t)j * fi + i] = (__bf16)W[idx];
}

__global__ void k_init_attn(float* __restrict__ amax, float* __restrict__ denom, int n) {
    int i = blockIdx.x * blockDim.x + threadIdx.x;
    if (i < n) { amax[i] = -__builtin_inff(); denom[i] = 0.0f; }
}

// ---------------------------------------------------------------------------
// Attention kernels (memory/atomic bound) — float4-vectorized gathers
// ---------------------------------------------------------------------------
__device__ inline void atomicMaxF(float* addr, float val) {
    if (val >= 0.0f) atomicMax((int*)addr, __float_as_int(val));
    else             atomicMin((unsigned int*)addr, (unsigned int)__float_as_int(val));
}

__global__ void k_alpha(const int* __restrict__ src, const int* __restrict__ dst,
                        const float* __restrict__ Q, const float* __restrict__ Kb,
                        const __bf16* __restrict__ eb,
                        float* __restrict__ alpha, float* __restrict__ amax,
                        int E, int hd, int d, float inv_sqrt_d)
{
    int idx = blockIdx.x * blockDim.x + threadIdx.x;
    if (idx >= E * H_HEADS) return;
    int e = idx >> 2, h = idx & 3;
    int s = src[e], t = dst[e];
    const float4* q4 = (const float4*)(Q  + (size_t)t * hd + h * d);
    const float4* k4 = (const float4*)(Kb + (size_t)s * hd + h * d);
    const v4bf*   e4 = (const v4bf*)  (eb + (size_t)e * hd + h * d);
    float acc = 0.0f;
    for (int j = 0; j < d / 4; ++j) {
        float4 qv = q4[j], kv = k4[j];
        v4bf   ev = e4[j];
        acc += qv.x * (kv.x + (float)ev[0]);
        acc += qv.y * (kv.y + (float)ev[1]);
        acc += qv.z * (kv.z + (float)ev[2]);
        acc += qv.w * (kv.w + (float)ev[3]);
    }
    float a = acc * inv_sqrt_d;
    alpha[idx] = a;
    atomicMaxF(&amax[(size_t)t * H_HEADS + h], a);
}

__global__ void k_expsum(const int* __restrict__ dst, float* __restrict__ alpha,
                         const float* __restrict__ amax, float* __restrict__ denom, int E)
{
    int idx = blockIdx.x * blockDim.x + threadIdx.x;
    if (idx >= E * H_HEADS) return;
    int e = idx >> 2, h = idx & 3;
    int t = dst[e];
    float ex = __expf(alpha[idx] - amax[(size_t)t * H_HEADS + h]);
    alpha[idx] = ex;
    atomicAdd(&denom[(size_t)t * H_HEADS + h], ex);
}

__global__ void k_scatter(const int* __restrict__ src, const int* __restrict__ dst,
                          const float* __restrict__ V, const __bf16* __restrict__ eb,
                          const float* __restrict__ alpha, const float* __restrict__ denom,
                          float* __restrict__ out, int E, int hd, int d)
{
    int idx = blockIdx.x * blockDim.x + threadIdx.x;
    if (idx >= E * H_HEADS) return;
    int e = idx >> 2, h = idx & 3;
    int s = src[e], t = dst[e];
    float attn = alpha[idx] / (denom[(size_t)t * H_HEADS + h] + 1e-16f);
    const float4* v4 = (const float4*)(V  + (size_t)s * hd + h * d);
    const v4bf*   e4 = (const v4bf*)  (eb + (size_t)e * hd + h * d);
    float* o = out + (size_t)t * hd + h * d;
    for (int j = 0; j < d / 4; ++j) {
        float4 vv = v4[j];
        v4bf   ev = e4[j];
        atomicAdd(&o[4 * j + 0], (vv.x + (float)ev[0]) * attn);
        atomicAdd(&o[4 * j + 1], (vv.y + (float)ev[1]) * attn);
        atomicAdd(&o[4 * j + 2], (vv.z + (float)ev[2]) * attn);
        atomicAdd(&o[4 * j + 3], (vv.w + (float)ev[3]) * attn);
    }
}

// ---------------------------------------------------------------------------
// BatchNorm (biased var, matches x.var(0)) + ReLU + bf16 convert
// ---------------------------------------------------------------------------
__global__ __launch_bounds__(256) void k_bn_stats(
    const float* __restrict__ x, const float* __restrict__ gamma,
    const float* __restrict__ beta, float* __restrict__ scale,
    float* __restrict__ shift, int n, int c)
{
    int ch = blockIdx.x;
    __shared__ float s1[256], s2[256];
    float a = 0.0f, b = 0.0f;
    for (int i = threadIdx.x; i < n; i += 256) {
        float v = x[(size_t)i * c + ch];
        a += v; b += v * v;
    }
    s1[threadIdx.x] = a; s2[threadIdx.x] = b;
    __syncthreads();
    for (int s = 128; s > 0; s >>= 1) {
        if ((int)threadIdx.x < s) {
            s1[threadIdx.x] += s1[threadIdx.x + s];
            s2[threadIdx.x] += s2[threadIdx.x + s];
        }
        __syncthreads();
    }
    if (threadIdx.x == 0) {
        float m   = s1[0] / (float)n;
        float var = s2[0] / (float)n - m * m;
        float sc  = gamma[ch] * rsqrtf(var + 1e-5f);
        scale[ch] = sc;
        shift[ch] = beta[ch] - m * sc;
    }
}

__global__ void k_bn_relu_bf16(const float* __restrict__ x, const float* __restrict__ scale,
                               const float* __restrict__ shift, __bf16* __restrict__ out,
                               size_t n, int c)
{
    size_t idx = (size_t)blockIdx.x * blockDim.x + threadIdx.x;
    if (idx >= n) return;
    int ch = (int)(idx % c);
    float v = x[idx] * scale[ch] + shift[ch];
    out[idx] = (__bf16)fmaxf(v, 0.0f);
}

// ---------------------------------------------------------------------------
// Host orchestration
// ---------------------------------------------------------------------------
static inline void launch_gemm(const __bf16* A, const __bf16* Bt, const float* bias,
                               float* Cf, __bf16* Cb, int M, int K, int fo,
                               hipStream_t stream)
{
    dim3 blk(256);
    dim3 grid((unsigned)((M / 16 + 7) / 8));
    if (fo == 128) {
        if (K == 128) k_gemm_bf16_wmma<8, 128><<<grid, blk, 0, stream>>>(A, Bt, bias, Cf, Cb, M);
        else          k_gemm_bf16_wmma<8,  64><<<grid, blk, 0, stream>>>(A, Bt, bias, Cf, Cb, M);
    } else {
        if (K == 128) k_gemm_bf16_wmma<4, 128><<<grid, blk, 0, stream>>>(A, Bt, bias, Cf, Cb, M);
        else          k_gemm_bf16_wmma<4,  64><<<grid, blk, 0, stream>>>(A, Bt, bias, Cf, Cb, M);
    }
}

extern "C" void kernel_launch(void* const* d_in, const int* in_sizes, int n_in,
                              void* d_out, int out_size, void* d_ws, size_t ws_size,
                              hipStream_t stream)
{
    (void)n_in; (void)out_size; (void)ws_size;
    const int IN = 64, HID = 128, OUTD = 64, H = H_HEADS;
    const int N = in_sizes[0] / IN;
    const int E = in_sizes[1] / 2;

    const float* x0  = (const float*)d_in[0];
    const int*   adj = (const int*)d_in[1];
    const int*   src = adj;
    const int*   dst = adj + E;
    const float* ea  = (const float*)d_in[2];
    const float* gamma[2] = { (const float*)d_in[30], (const float*)d_in[32] };
    const float* beta[2]  = { (const float*)d_in[31], (const float*)d_in[33] };

    char* p = (char*)d_ws;
    auto take = [&](size_t bytes) -> void* {
        void* r = (void*)p;
        p += (bytes + 255) & ~(size_t)255;
        return r;
    };

    __bf16* ea_bf  = (__bf16*)take((size_t)E * IN * 2);
    __bf16* x_bf   = (__bf16*)take((size_t)N * HID * 2);
    float*  Qb     = (float*)take((size_t)N * HID * 4);
    float*  Kb     = (float*)take((size_t)N * HID * 4);
    float*  Vb     = (float*)take((size_t)N * HID * 4);
    float*  OUTb   = (float*)take((size_t)N * HID * 4);
    __bf16* ebuf   = (__bf16*)take((size_t)E * HID * 2);
    float*  alphab = (float*)take((size_t)E * H * 4);
    float*  amax   = (float*)take((size_t)N * H * 4);
    float*  denom  = (float*)take((size_t)N * H * 4);
    float*  scale  = (float*)take(HID * 4);
    float*  shift  = (float*)take(HID * 4);
    __bf16* wtpool = (__bf16*)take(262144 * 2);

    // Convert activations / edge features to bf16 once.
    {
        size_t n1 = (size_t)N * IN;
        k_f32_to_bf16<<<(unsigned)((n1 + 255) / 256), 256, 0, stream>>>(x0, x_bf, n1);
        size_t n2 = (size_t)E * IN;
        k_f32_to_bf16<<<(unsigned)((n2 + 255) / 256), 256, 0, stream>>>(ea, ea_bf, n2);
    }

    // Transpose all weights into bf16 [fo x fi] fragment-friendly layout.
    __bf16 *WqT[3], *WkT[3], *WvT[3], *WeT[3], *WsT[3];
    const float *bq[3], *bk[3], *bv[3], *bs[3];
    size_t woff = 0;
    for (int l = 0; l < 3; ++l) {
        const int fi = l ? HID : IN;
        const int fo = (l == 2) ? OUTD : HID;
        const float* Wq = (const float*)d_in[3 + 9 * l + 0];
        bq[l]           = (const float*)d_in[3 + 9 * l + 1];
        const float* Wk = (const float*)d_in[3 + 9 * l + 2];
        bk[l]           = (const float*)d_in[3 + 9 * l + 3];
        const float* Wv = (const float*)d_in[3 + 9 * l + 4];
        bv[l]           = (const float*)d_in[3 + 9 * l + 5];
        const float* We = (const float*)d_in[3 + 9 * l + 6];
        const float* Ws = (const float*)d_in[3 + 9 * l + 7];
        bs[l]           = (const float*)d_in[3 + 9 * l + 8];

        auto tr = [&](const float* W, int a, int b) -> __bf16* {
            __bf16* dstw = wtpool + woff;
            woff += (size_t)a * b;
            k_transpose_bf16<<<(a * b + 255) / 256, 256, 0, stream>>>(W, dstw, a, b);
            return dstw;
        };
        WqT[l] = tr(Wq, fi, fo);
        WkT[l] = tr(Wk, fi, fo);
        WvT[l] = tr(Wv, fi, fo);
        WeT[l] = tr(We, IN, fo);
        WsT[l] = tr(Ws, fi, fo);
    }

    const int EH = E * H;
    for (int l = 0; l < 3; ++l) {
        const int fi = l ? HID : IN;
        const int fo = (l == 2) ? OUTD : HID;
        const int d  = fo / H;
        float* outp = (l == 2) ? (float*)d_out : OUTb;

        launch_gemm(x_bf, WqT[l], bq[l], Qb, nullptr, N, fi, fo, stream);
        launch_gemm(x_bf, WkT[l], bk[l], Kb, nullptr, N, fi, fo, stream);
        launch_gemm(x_bf, WvT[l], bv[l], Vb, nullptr, N, fi, fo, stream);
        // skip connection initializes the output accumulator
        launch_gemm(x_bf, WsT[l], bs[l], outp, nullptr, N, fi, fo, stream);
        // edge features (bf16 output, no bias)
        launch_gemm(ea_bf, WeT[l], nullptr, nullptr, ebuf, E, IN, fo, stream);

        k_init_attn<<<(N * H + 255) / 256, 256, 0, stream>>>(amax, denom, N * H);
        k_alpha  <<<(EH + 255) / 256, 256, 0, stream>>>(src, dst, Qb, Kb, ebuf, alphab, amax,
                                                        E, fo, d, 1.0f / sqrtf((float)d));
        k_expsum <<<(EH + 255) / 256, 256, 0, stream>>>(dst, alphab, amax, denom, E);
        k_scatter<<<(EH + 255) / 256, 256, 0, stream>>>(src, dst, Vb, ebuf, alphab, denom,
                                                        outp, E, fo, d);

        if (l < 2) {
            k_bn_stats<<<fo, 256, 0, stream>>>(OUTb, gamma[l], beta[l], scale, shift, N, fo);
            size_t nt = (size_t)N * fo;
            k_bn_relu_bf16<<<(unsigned)((nt + 255) / 256), 256, 0, stream>>>(
                OUTb, scale, shift, x_bf, nt, fo);
        }
    }

    // Second tuple output: edge_attr passthrough.
    hipMemcpyAsync((float*)d_out + (size_t)N * OUTD, ea, (size_t)E * IN * 4,
                   hipMemcpyDeviceToDevice, stream);
}
